// MultiHeadAttention_2413771620962
// MI455X (gfx1250) — matrix operand order repaired and checked by reading, not compiled, not run on hardware
//
#include <hip/hip_runtime.h>
#include <hip/hip_bf16.h>

// ---------------------------------------------------------------------------
// MHA block for MI455X (gfx1250, wave32, WMMA).
// Precision: bf16 A/B fragments, fp32 accumulate via v_wmma_f32_16x16x32_bf16.
// V is stored head-split TRANSPOSED so the P@V B-fragments are contiguous.
// Q tiles staged to LDS via global_load_async_to_lds_b128 (ASYNCcnt path).
// ---------------------------------------------------------------------------

#define D_MODEL  1024
#define NUM_HEADS 16
#define D_K       64
#define SEQ     2048
#define BATCH      2
#define M_ROWS  (BATCH * SEQ)      // 4096
#define LN_EPS  1e-6f

// K2 LDS layout (bytes):
//   [0, 131072)        : fp32 score strip [16][2048]
//   [131072, 163840)   : fp32 PV partials [8][16][64]
//   [163840, 165888)   : bf16 Q tile [16][64]
#define SC_FLOATS    (16 * SEQ)
#define PART_FLOATS  (8 * 16 * 64)
#define QTILE_OFF_B  ((SC_FLOATS + PART_FLOATS) * 4)
#define K2_LDS_BYTES (QTILE_OFF_B + 16 * D_K * 2)

typedef __bf16 bf16;
typedef __attribute__((ext_vector_type(16))) __bf16 v16bf;
typedef __attribute__((ext_vector_type(8)))  float  v8f;

static __device__ __forceinline__ v8f wmma_bf16(v16bf a, v16bf b, v8f c) {
  // (neg_a, A, neg_b, B, c_mod, C, reuse_a, reuse_b)
  return __builtin_amdgcn_wmma_f32_16x16x32_bf16(false, a, false, b,
                                                 (short)0, c, false, false);
}

// ---- Fragment loaders (layouts per CDNA5 ISA 7.12.2, wave32) --------------
// A (16x32, 16-bit): lane L holds row m=L%16; lanes 0-15 K={0..7,16..23},
// lanes 16-31 K={8..15,24..31}; halves j<8 -> K=koff+j, j>=8 -> K=koff+16+(j-8).
static __device__ __forceinline__ v16bf frag_a_f32(const float* __restrict__ X,
                                                   int ld, int m0, int k0, int lane) {
  int m    = m0 + (lane & 15);
  int koff = ((lane >> 4) & 1) * 8;
  const float* p = X + (size_t)m * ld + k0 + koff;
  v16bf a;
#pragma unroll
  for (int j = 0; j < 8; ++j) a[j] = (bf16)p[j];
#pragma unroll
  for (int j = 0; j < 8; ++j) a[8 + j] = (bf16)p[16 + j];
  return a;
}

static __device__ __forceinline__ v16bf frag_a_bf16(const bf16* X,
                                                    int ld, int m0, int k0, int lane) {
  int m    = m0 + (lane & 15);
  int koff = ((lane >> 4) & 1) * 8;
  const bf16* p = X + (size_t)m * ld + k0 + koff;
  v16bf a;
#pragma unroll
  for (int j = 0; j < 8; ++j) a[j] = p[j];
#pragma unroll
  for (int j = 0; j < 8; ++j) a[8 + j] = p[16 + j];
  return a;
}

// A fragment out of an fp32 LDS strip (softmax probabilities)
static __device__ __forceinline__ v16bf frag_a_lds_f32(const float* S, int ld,
                                                       int k0, int lane) {
  int m    = lane & 15;
  int koff = ((lane >> 4) & 1) * 8;
  const float* p = S + m * ld + k0 + koff;
  v16bf a;
#pragma unroll
  for (int j = 0; j < 8; ++j) a[j] = (bf16)p[j];
#pragma unroll
  for (int j = 0; j < 8; ++j) a[8 + j] = (bf16)p[16 + j];
  return a;
}

// B (32x16, 16-bit): lane L holds column n=L%16; lanes 0-15 K={0..7,16..23},
// lanes 16-31 K={8..15,24..31} (same order as A).  For B[k][n] = W[n][k]
// (row-major W over k) this is two 8-element runs of row n.
static __device__ __forceinline__ v16bf frag_b_wT_f32(const float* __restrict__ W,
                                                      int ld, int k0, int n0, int lane) {
  int n    = n0 + (lane & 15);
  int koff = ((lane >> 4) & 1) * 8;
  const float* p = W + (size_t)n * ld + k0 + koff;
  v16bf b;
#pragma unroll
  for (int j = 0; j < 8; ++j) b[j] = (bf16)p[j];
#pragma unroll
  for (int j = 0; j < 8; ++j) b[8 + j] = (bf16)p[16 + j];
  return b;
}

static __device__ __forceinline__ v16bf frag_b_wT_bf16(const bf16* __restrict__ W,
                                                       int ld, int k0, int n0, int lane) {
  int n    = n0 + (lane & 15);
  int koff = ((lane >> 4) & 1) * 8;
  const bf16* p = W + (size_t)n * ld + k0 + koff;
  v16bf b;
#pragma unroll
  for (int j = 0; j < 8; ++j) b[j] = p[j];
#pragma unroll
  for (int j = 0; j < 8; ++j) b[8 + j] = p[16 + j];
  return b;
}

// ---------------------------------------------------------------------------
// Kernel 1: fused Q/K/V projection.  y = x @ W^T + b.
// qh, kh written head-split [B,H,S,dk]; vh written head-split TRANSPOSED
// [B,H,dk,S] so P@V B-fragments are contiguous.  One wave: 16(M) x 64(N).
// ---------------------------------------------------------------------------
__global__ void qkv_proj_kernel(const float* __restrict__ q,
                                const float* __restrict__ k,
                                const float* __restrict__ v,
                                const float* __restrict__ Wq, const float* __restrict__ bq,
                                const float* __restrict__ Wk, const float* __restrict__ bk,
                                const float* __restrict__ Wv, const float* __restrict__ bv,
                                bf16* __restrict__ qh, bf16* __restrict__ kh,
                                bf16* __restrict__ vh) {
  int wave = threadIdx.x >> 5;
  int lane = threadIdx.x & 31;
  int task = blockIdx.x * 4 + wave;      // 3 * 256 * 16 = 12288 tasks
  int mat  = task >> 12;                 // 0..2
  int rem  = task & 4095;
  int m0   = (rem >> 4) << 4;            // row tile * 16
  int ng   = rem & 15;                   // 64-wide N group

  const float* X; const float* W; const float* bias; bf16* dst;
  if (mat == 0)      { X = q; W = Wq; bias = bq; dst = qh; }
  else if (mat == 1) { X = k; W = Wk; bias = bk; dst = kh; }
  else               { X = v; W = Wv; bias = bv; dst = vh; }

  v8f acc[4] = {};
  for (int kk = 0; kk < D_MODEL; kk += 32) {
    // prefetch next A-row chunk (global_prefetch_b8)
    __builtin_prefetch(X + (size_t)(m0 + (lane & 15)) * D_MODEL + kk + 32, 0, 3);
    v16bf a = frag_a_f32(X, D_MODEL, m0, kk, lane);
#pragma unroll
    for (int t = 0; t < 4; ++t) {
      v16bf b = frag_b_wT_f32(W, D_MODEL, kk, ng * 64 + t * 16, lane);
      acc[t] = wmma_bf16(a, b, acc[t]);
    }
  }

  int mrow = m0 + ((lane >> 4) & 1) * 8;
#pragma unroll
  for (int t = 0; t < 4; ++t) {
    int n  = ng * 64 + t * 16 + (lane & 15);
    int h  = n >> 6;
    int dk = n & 63;
#pragma unroll
    for (int r = 0; r < 8; ++r) {
      int m  = mrow + r;
      int bb = m >> 11;         // / SEQ
      int s  = m & 2047;
      float val = acc[t][r] + bias[n];
      size_t idx;
      if (mat == 2)  // vh: [B,H,dk,S]
        idx = ((size_t)(bb * NUM_HEADS + h) * D_K + dk) * SEQ + s;
      else           // qh/kh: [B,H,S,dk]
        idx = ((size_t)(bb * NUM_HEADS + h) * SEQ + s) * D_K + dk;
      dst[idx] = (bf16)val;
    }
  }
}

// ---------------------------------------------------------------------------
// Kernel 2: scores (QK^T * scale, mask) -> softmax (LDS + wave32 shuffles,
// attn probs written to d_out) -> PV (wave-sliced K, LDS partial reduce)
// -> ctx written bf16 to [M_ROWS, D_MODEL].
// Q tile staged to LDS via CDNA5 async copy.
// ---------------------------------------------------------------------------
__global__ void attention_kernel(const bf16* __restrict__ qh,
                                 const bf16* __restrict__ kh,
                                 const bf16* __restrict__ vh,   // [B,H,dk,S]
                                 const int*  __restrict__ mask,
                                 float* __restrict__ attn_out,
                                 bf16*  __restrict__ ctx) {
  extern __shared__ float smem[];
  float* sc   = smem;                     // [16][2048]
  float* part = smem + SC_FLOATS;         // [8][16][64]
  const bf16* qtile = (const bf16*)((const char*)smem + QTILE_OFF_B); // [16][64]

  int bh = blockIdx.x;              // 0..31  (b*16 + h)
  int qt = blockIdx.y;              // 0..127
  int b  = bh >> 4;
  int h  = bh & 15;
  int wave = threadIdx.x >> 5;
  int lane = threadIdx.x & 31;
  int q0   = qt * 16;

  const bf16* qH  = qh + (size_t)bh * SEQ * D_K;
  const bf16* kH  = kh + (size_t)bh * SEQ * D_K;
  const bf16* vHT = vh + (size_t)bh * D_K * SEQ;   // [64][2048]
  const int*  mk  = mask + b * SEQ;
  const float scale = 0.125f;       // 1/sqrt(64)

  // ---- stage 16x64 bf16 Q tile (2 KB, contiguous) into LDS, async path ----
  if (threadIdx.x < 128) {
    unsigned lds_off = (unsigned)QTILE_OFF_B + threadIdx.x * 16u;
    const char* gsrc = (const char*)(qH + (size_t)q0 * D_K) + threadIdx.x * 16u;
    asm volatile("global_load_async_to_lds_b128 %0, %1, off"
                 :: "v"(lds_off), "v"(gsrc) : "memory");
  }
  asm volatile("s_wait_asynccnt 0x0" ::: "memory");
  __syncthreads();

  // ---- phase 1: scores -> LDS strip ----
  v16bf aq0 = frag_a_bf16(qtile, D_K, 0, 0,  lane);
  v16bf aq1 = frag_a_bf16(qtile, D_K, 0, 32, lane);
  for (int kt = wave; kt < SEQ / 16; kt += 8) {
    v8f acc = {};
    v16bf b0 = frag_b_wT_bf16(kH, D_K, 0,  kt * 16, lane);
    acc = wmma_bf16(aq0, b0, acc);
    v16bf b1 = frag_b_wT_bf16(kH, D_K, 32, kt * 16, lane);
    acc = wmma_bf16(aq1, b1, acc);

    int col   = kt * 16 + (lane & 15);
    int msk   = mk[col];
    int rbase = ((lane >> 4) & 1) * 8;
#pragma unroll
    for (int r = 0; r < 8; ++r) {
      float s = acc[r] * scale;
      if (msk == 0) s = -1e9f;
      sc[(rbase + r) * SEQ + col] = s;
    }
  }
  __syncthreads();

  // ---- phase 2: row softmax (2 rows per wave), write attn probs out ----
#pragma unroll
  for (int rr = 0; rr < 2; ++rr) {
    int row = wave * 2 + rr;
    float* srow = sc + row * SEQ;
    float mx = -3.4e38f;
    for (int c = lane; c < SEQ; c += 32) mx = fmaxf(mx, srow[c]);
#pragma unroll
    for (int off = 16; off > 0; off >>= 1) mx = fmaxf(mx, __shfl_xor(mx, off, 32));
    float sum = 0.f;
    for (int c = lane; c < SEQ; c += 32) {
      float e = __expf(srow[c] - mx);
      srow[c] = e;
      sum += e;
    }
#pragma unroll
    for (int off = 16; off > 0; off >>= 1) sum += __shfl_xor(sum, off, 32);
    float inv = 1.0f / sum;
    float* arow = attn_out + ((size_t)bh * SEQ + q0 + row) * SEQ;
    for (int c = lane; c < SEQ; c += 32) {
      float p = srow[c] * inv;
      srow[c] = p;
      arow[c] = p;
    }
  }
  __syncthreads();

  // ---- phase 3: ctx = P @ V, wave w covers K in [w*256, w*256+256) ----
  // B[k][n=dk] = V[k,dk] = vHT[dk*SEQ + k]  -> contiguous per lane.
  v8f cacc[4] = {};
  for (int kk = wave * 256; kk < wave * 256 + 256; kk += 32) {
    v16bf a = frag_a_lds_f32(sc, SEQ, kk, lane);
#pragma unroll
    for (int t = 0; t < 4; ++t) {
      v16bf bb = frag_b_wT_bf16(vHT, SEQ, kk, t * 16, lane);
      cacc[t] = wmma_bf16(a, bb, cacc[t]);
    }
  }
  {
    int rbase = ((lane >> 4) & 1) * 8;
#pragma unroll
    for (int t = 0; t < 4; ++t) {
      int n = t * 16 + (lane & 15);
#pragma unroll
      for (int r = 0; r < 8; ++r)
        part[(wave * 16 + rbase + r) * 64 + n] = cacc[t][r];
    }
  }
  __syncthreads();

  for (int idx = threadIdx.x; idx < 16 * 64; idx += blockDim.x) {
    int row = idx >> 6;
    int col = idx & 63;
    float s = 0.f;
#pragma unroll
    for (int w = 0; w < 8; ++w) s += part[(w * 16 + row) * 64 + col];
    int m = b * SEQ + q0 + row;
    ctx[(size_t)m * D_MODEL + h * D_K + col] = (bf16)s;
  }
}

// ---------------------------------------------------------------------------
// Kernel 3: out = ctx @ Wo^T + bo + residual  (fp32 into workspace for LN)
// ---------------------------------------------------------------------------
__global__ void out_proj_kernel(const bf16* __restrict__ ctx,
                                const float* __restrict__ Wo,
                                const float* __restrict__ bo,
                                const float* __restrict__ resid,
                                float* __restrict__ xout) {
  int wave = threadIdx.x >> 5;
  int lane = threadIdx.x & 31;
  int task = blockIdx.x * 4 + wave;     // 256 * 16 = 4096 tasks
  int m0   = (task >> 4) << 4;
  int ng   = task & 15;

  v8f acc[4] = {};
  for (int kk = 0; kk < D_MODEL; kk += 32) {
    __builtin_prefetch(ctx + (size_t)(m0 + (lane & 15)) * D_MODEL + kk + 32, 0, 3);
    v16bf a = frag_a_bf16(ctx, D_MODEL, m0, kk, lane);
#pragma unroll
    for (int t = 0; t < 4; ++t) {
      v16bf b = frag_b_wT_f32(Wo, D_MODEL, kk, ng * 64 + t * 16, lane);
      acc[t] = wmma_bf16(a, b, acc[t]);
    }
  }

  int mrow = m0 + ((lane >> 4) & 1) * 8;
#pragma unroll
  for (int t = 0; t < 4; ++t) {
    int n = ng * 64 + t * 16 + (lane & 15);
#pragma unroll
    for (int r = 0; r < 8; ++r) {
      int m = mrow + r;
      float val = acc[t][r] + bo[n] + resid[(size_t)m * D_MODEL + n];
      xout[(size_t)m * D_MODEL + n] = val;
    }
  }
}

// ---------------------------------------------------------------------------
// Kernel 4: row LayerNorm -> y region of d_out
// ---------------------------------------------------------------------------
__global__ void layernorm_kernel(const float* __restrict__ x,
                                 const float* __restrict__ gamma,
                                 const float* __restrict__ beta,
                                 float* __restrict__ y) {
  __shared__ float sbuf[8];
  int row  = blockIdx.x;
  int lane = threadIdx.x & 31;
  int wave = threadIdx.x >> 5;
  const float* xr = x + (size_t)row * D_MODEL;

  float s = 0.f;
  for (int c = threadIdx.x; c < D_MODEL; c += 256) s += xr[c];
#pragma unroll
  for (int off = 16; off > 0; off >>= 1) s += __shfl_xor(s, off, 32);
  if (lane == 0) sbuf[wave] = s;
  __syncthreads();
  float mean = 0.f;
#pragma unroll
  for (int w = 0; w < 8; ++w) mean += sbuf[w];
  mean *= (1.0f / D_MODEL);
  __syncthreads();

  float vs = 0.f;
  for (int c = threadIdx.x; c < D_MODEL; c += 256) {
    float d = xr[c] - mean;
    vs += d * d;
  }
#pragma unroll
  for (int off = 16; off > 0; off >>= 1) vs += __shfl_xor(vs, off, 32);
  if (lane == 0) sbuf[wave] = vs;
  __syncthreads();
  float var = 0.f;
#pragma unroll
  for (int w = 0; w < 8; ++w) var += sbuf[w];
  var *= (1.0f / D_MODEL);
  float rstd = rsqrtf(var + LN_EPS);

  float* yr = y + (size_t)row * D_MODEL;
  for (int c = threadIdx.x; c < D_MODEL; c += 256)
    yr[c] = (xr[c] - mean) * rstd * gamma[c] + beta[c];
}

// ---------------------------------------------------------------------------
// Launch.  Workspace layout (bytes):
//   [0,8M)    qh  bf16 [B,H,S,dk]
//   [8M,16M)  kh  bf16 [B,H,S,dk]
//   [16M,24M) vh  bf16 [B,H,dk,S]   (transposed)
//   [24M,32M) ctx bf16 [M_ROWS, D_MODEL]
//   [32M,48M) x   fp32 [M_ROWS, D_MODEL] (pre-LN)
// d_out: y fp32 (4096*1024) followed by attn fp32 (2*16*2048*2048).
// ---------------------------------------------------------------------------
extern "C" void kernel_launch(void* const* d_in, const int* in_sizes, int n_in,
                              void* d_out, int out_size, void* d_ws, size_t ws_size,
                              hipStream_t stream) {
  const float* q    = (const float*)d_in[0];
  const float* k    = (const float*)d_in[1];
  const float* v    = (const float*)d_in[2];
  const int*   mask = (const int*)  d_in[3];
  const float* Wq   = (const float*)d_in[4];
  const float* bq   = (const float*)d_in[5];
  const float* Wk   = (const float*)d_in[6];
  const float* bk   = (const float*)d_in[7];
  const float* Wv   = (const float*)d_in[8];
  const float* bv   = (const float*)d_in[9];
  const float* Wo   = (const float*)d_in[10];
  const float* bo   = (const float*)d_in[11];
  const float* gam  = (const float*)d_in[12];
  const float* bet  = (const float*)d_in[13];

  float* y    = (float*)d_out;
  float* attn = y + (size_t)M_ROWS * D_MODEL;

  char* ws = (char*)d_ws;
  const size_t MB8 = 8ull << 20;
  bf16*  qh   = (bf16*)(ws);
  bf16*  kh   = (bf16*)(ws + 1 * MB8);
  bf16*  vh   = (bf16*)(ws + 2 * MB8);
  bf16*  ctx  = (bf16*)(ws + 3 * MB8);
  float* xpre = (float*)(ws + 4 * MB8);

  // K1: 12288 wave-tasks / 4 waves per block
  qkv_proj_kernel<<<3072, 128, 0, stream>>>(q, k, v, Wq, bq, Wk, bk, Wv, bv,
                                            qh, kh, vh);

  // K2: one block per (b*h, q-tile); ~162 KB dynamic LDS
  dim3 g2(BATCH * NUM_HEADS, SEQ / 16);
  attention_kernel<<<g2, 256, K2_LDS_BYTES, stream>>>(qh, kh, vh, mask, attn, ctx);

  // K3: 4096 wave-tasks / 4 waves per block
  out_proj_kernel<<<1024, 128, 0, stream>>>(ctx, Wo, bo, q, xpre);

  // K4: one block per row
  layernorm_kernel<<<M_ROWS, 256, 0, stream>>>(xpre, gam, bet, y);
}
